// ESN_46282567582050
// MI455X (gfx1250) — matrix operand order repaired and checked by reading, not compiled, not run on hardware
//
#include <hip/hip_runtime.h>
#include <hip/hip_bf16.h>

typedef __attribute__((ext_vector_type(16))) _Float16 v16h;
typedef __attribute__((ext_vector_type(8)))  float    v8f;

#define ESN_T     16384
#define ESN_N     2048
#define NBLK      32                 // persistent workgroups (grid-barrier participants)
#define WAVES_PB  4                  // waves per workgroup (128 threads, wave32)
#define KCHUNKS   (ESN_N / 32)       // 64 K-chunks of 32 per matvec

// ---- workspace layout (d_ws) ----
// [0, 8MB)            : W packed to f16 in WMMA A-operand tile layout
// [8MB, +8KB)         : x state, f16, double buffered (2 * 2048 halves)
// [.., +256B)         : grid barrier {counter, generation}
// [..)                : per-(t, wg) readout partials, 16384*32*8 f32 = 16MB
#define WS_WPACK_BYTES ((size_t)ESN_N * ESN_N * 2)
#define WS_XBUF_OFF    (WS_WPACK_BYTES)
#define WS_XBUF_BYTES  ((size_t)2 * ESN_N * 2)
#define WS_BAR_OFF     (WS_XBUF_OFF + WS_XBUF_BYTES)
#define WS_BAR_BYTES   ((size_t)256)
#define WS_PART_OFF    (WS_BAR_OFF + WS_BAR_BYTES)

// ---------------------------------------------------------------------------
// Pack W (f32 row-major 2048x2048) into f16 tiles for the CDNA5 WMMA
// 16-bit A operand (16x32), in contiguous k order shared with B:
//   lane l (M = l&15), halves h0..15 -> K = (l>=16?16:0) + h
// Linearized so each lane in the scan kernel loads one contiguous v16h (32B).
// ---------------------------------------------------------------------------
__global__ void esn_pack_w(const float* __restrict__ W, _Float16* __restrict__ Wp) {
    int id = blockIdx.x * blockDim.x + threadIdx.x;     // 0 .. 2048*2048-1
    if (id >= ESN_N * ESN_N) return;
    int h = id & 15;
    int l = (id >> 4) & 31;
    int c = (id >> 9) & 63;          // K-chunk
    int r = id >> 15;                // 16-row tile
    int m = r * 16 + (l & 15);
    int k = c * 32 + h + ((l >> 4) << 4);
    Wp[id] = (_Float16)W[(size_t)m * ESN_N + k];
}

__global__ void esn_scan_init(_Float16* __restrict__ xbuf, int* __restrict__ bar) {
    int id = blockIdx.x * blockDim.x + threadIdx.x;
    if (id < 2 * ESN_N) xbuf[id] = (_Float16)0.0f;      // x0 = 0, both buffers
    if (id < 2) bar[id] = 0;                             // counter, generation
}

// Sense-reversing grid barrier over NBLK resident workgroups.
__device__ inline void grid_barrier(int* bar) {
    __threadfence();
    __syncthreads();
    if (threadIdx.x == 0) {
        int g = __hip_atomic_load(&bar[1], __ATOMIC_ACQUIRE, __HIP_MEMORY_SCOPE_AGENT);
        int a = __hip_atomic_fetch_add(&bar[0], 1, __ATOMIC_ACQ_REL, __HIP_MEMORY_SCOPE_AGENT);
        if (a == NBLK - 1) {
            __hip_atomic_store(&bar[0], 0, __ATOMIC_RELAXED, __HIP_MEMORY_SCOPE_AGENT);
            __hip_atomic_fetch_add(&bar[1], 1, __ATOMIC_RELEASE, __HIP_MEMORY_SCOPE_AGENT);
        } else {
            while (__hip_atomic_load(&bar[1], __ATOMIC_ACQUIRE, __HIP_MEMORY_SCOPE_AGENT) == g) {
                __builtin_amdgcn_s_sleep(1);
            }
        }
    }
    __syncthreads();
}

// ---------------------------------------------------------------------------
// Persistent scan: each wave owns one 16-row tile of W (tile = blk*4 + wave).
// Per step: stage x (f16) into LDS, 64 chained WMMAs over K (4 interleaved
// accumulators), then lanes 0/16 (which hold D[:,0] = y rows 0-7 / 8-15)
// apply u_proj + bias + tanh + leak, publish x_next (f16) and readout partials.
// ---------------------------------------------------------------------------
__global__ void __launch_bounds__(128, 1)
esn_scan(const float* __restrict__ inp,   // (T, 8)
         const float* __restrict__ Win,   // (N, 8)
         const float* __restrict__ bvec,  // (N)
         const float* __restrict__ Wout,  // (8, N+1)
         const _Float16* __restrict__ Wpack,
         _Float16* __restrict__ xbuf,     // 2 * N halves
         int* __restrict__ bar,
         float* __restrict__ partial)     // (T, NBLK, 8)
{
    __shared__ __align__(32) _Float16 xs[ESN_N];   // 4 KB staged x
    __shared__ float red[WAVES_PB * 2 * 8];        // per-half-wave readout partials

    const int tid  = threadIdx.x;
    const int lane = tid & 31;
    const int wave = tid >> 5;
    const int tile = blockIdx.x * WAVES_PB + wave; // 0..127
    const int m0   = tile * 16;

    const bool owner = (lane == 0) || (lane == 16);
    const int  rbase = m0 + ((lane >= 16) ? 8 : 0);
    const int  koff  = (lane >= 16) ? 16 : 0;
    const _Float16* wpt = Wpack + (size_t)tile * (KCHUNKS * 512);

    // Per-row constants + recurrent state live in registers on lanes 0/16.
    float xstate[8], bb[8], win[8][8], wout[8][8];
    if (owner) {
#pragma unroll
        for (int j = 0; j < 8; ++j) {
            xstate[j] = 0.0f;
            bb[j] = bvec[rbase + j];
#pragma unroll
            for (int d = 0; d < 8; ++d) win[j][d] = Win[(rbase + j) * 8 + d];
#pragma unroll
            for (int o = 0; o < 8; ++o) wout[o][j] = Wout[o * (ESN_N + 1) + (rbase + j)];
        }
    }

    for (int t = 0; t < ESN_T; ++t) {
        const _Float16* xcur = xbuf + (size_t)(t & 1) * ESN_N;
        _Float16*       xnxt = xbuf + (size_t)((t + 1) & 1) * ESN_N;

        // Stage x into LDS: 128 threads x 32B = 4KB.
        *(v16h*)(xs + tid * 16) = *(const v16h*)(xcur + tid * 16);
        __syncthreads();

        // y = W_tile @ x via WMMA; B = x-slice replicated across 16 columns.
        v8f acc0 = {}, acc1 = {}, acc2 = {}, acc3 = {};
#pragma unroll 1
        for (int c = 0; c < KCHUNKS; c += 4) {
            v16h a0 = *(const v16h*)(wpt + (size_t)(c + 0) * 512 + lane * 16);
            v16h a1 = *(const v16h*)(wpt + (size_t)(c + 1) * 512 + lane * 16);
            v16h a2 = *(const v16h*)(wpt + (size_t)(c + 2) * 512 + lane * 16);
            v16h a3 = *(const v16h*)(wpt + (size_t)(c + 3) * 512 + lane * 16);
            v16h b0 = *(const v16h*)(xs + (c + 0) * 32 + koff);
            v16h b1 = *(const v16h*)(xs + (c + 1) * 32 + koff);
            v16h b2 = *(const v16h*)(xs + (c + 2) * 32 + koff);
            v16h b3 = *(const v16h*)(xs + (c + 3) * 32 + koff);
            acc0 = __builtin_amdgcn_wmma_f32_16x16x32_f16(false, a0, false, b0, (short)0, acc0, false, false);
            acc1 = __builtin_amdgcn_wmma_f32_16x16x32_f16(false, a1, false, b1, (short)0, acc1, false, false);
            acc2 = __builtin_amdgcn_wmma_f32_16x16x32_f16(false, a2, false, b2, (short)0, acc2, false, false);
            acc3 = __builtin_amdgcn_wmma_f32_16x16x32_f16(false, a3, false, b3, (short)0, acc3, false, false);
        }
        v8f accT = acc0 + acc1;
        accT = accT + acc2;
        accT = accT + acc3;

        // Lane 0 holds y[m0..m0+7] in accT[0..7]; lane 16 holds y[m0+8..m0+15].
        if (owner) {
            const float4* ip4 = (const float4*)(inp + (size_t)t * 8);
            float4 ia = ip4[0], ib = ip4[1];
            float iv[8] = {ia.x, ia.y, ia.z, ia.w, ib.x, ib.y, ib.z, ib.w};
            float p[8];
#pragma unroll
            for (int o = 0; o < 8; ++o) p[o] = 0.0f;
#pragma unroll
            for (int j = 0; j < 8; ++j) {
                float u = bb[j] + accT[j];
#pragma unroll
                for (int d = 0; d < 8; ++d) u += win[j][d] * iv[d];
                float xn = 0.1f * xstate[j] + 0.9f * tanhf(u);
                xstate[j] = xn;
                xnxt[rbase + j] = (_Float16)xn;
#pragma unroll
                for (int o = 0; o < 8; ++o) p[o] += wout[o][j] * xn;
            }
            float* rr = &red[(wave * 2 + (lane >= 16 ? 1 : 0)) * 8];
#pragma unroll
            for (int o = 0; o < 8; ++o) rr[o] = p[o];
        }
        __syncthreads();
        if (tid < 8) {
            float s = 0.0f;
#pragma unroll
            for (int w2 = 0; w2 < WAVES_PB * 2; ++w2) s += red[w2 * 8 + tid];
            partial[((size_t)t * NBLK + blockIdx.x) * 8 + tid] = s;
        }
        grid_barrier(bar);   // publishes xnxt + partials, separates LDS reuse
    }
}

__global__ void esn_readout_reduce(const float* __restrict__ partial,
                                   const float* __restrict__ Wout,
                                   float* __restrict__ out) {
    int id = blockIdx.x * blockDim.x + threadIdx.x;    // 0 .. T*8-1
    if (id >= ESN_T * 8) return;
    int t = id >> 3, o = id & 7;
    float s = Wout[o * (ESN_N + 1) + ESN_N];           // bias column
#pragma unroll 4
    for (int wg = 0; wg < NBLK; ++wg) s += partial[((size_t)t * NBLK + wg) * 8 + o];
    out[id] = s;
}

extern "C" void kernel_launch(void* const* d_in, const int* in_sizes, int n_in,
                              void* d_out, int out_size, void* d_ws, size_t ws_size,
                              hipStream_t stream) {
    const float* inputs = (const float*)d_in[0];   // (16384, 8)
    const float* W_in   = (const float*)d_in[1];   // (2048, 8)
    const float* W      = (const float*)d_in[2];   // (2048, 2048)
    const float* b      = (const float*)d_in[3];   // (2048)
    const float* W_out  = (const float*)d_in[4];   // (8, 2049)
    float* out = (float*)d_out;

    char* ws = (char*)d_ws;
    _Float16* wpack   = (_Float16*)ws;
    _Float16* xbuf    = (_Float16*)(ws + WS_XBUF_OFF);
    int*      bar     = (int*)(ws + WS_BAR_OFF);
    float*    partial = (float*)(ws + WS_PART_OFF);

    esn_pack_w<<<(ESN_N * ESN_N + 255) / 256, 256, 0, stream>>>(W, wpack);
    esn_scan_init<<<(2 * ESN_N + 255) / 256, 256, 0, stream>>>(xbuf, bar);
    esn_scan<<<NBLK, WAVES_PB * 32, 0, stream>>>(inputs, W_in, b, W_out,
                                                 wpack, xbuf, bar, partial);
    esn_readout_reduce<<<(ESN_T * 8 + 255) / 256, 256, 0, stream>>>(partial, W_out, out);
}
